// CausalScaledDotAttention_42880953484349
// MI455X (gfx1250) — matrix operand order repaired and checked by reading, not compiled, not run on hardware
//
#include <hip/hip_runtime.h>
#include <hip/hip_bf16.h>

typedef __attribute__((ext_vector_type(16))) __bf16       v16bf;
typedef __attribute__((ext_vector_type(8)))  float        v8f;
typedef __attribute__((ext_vector_type(4))) unsigned int  u32x4;
typedef __attribute__((ext_vector_type(4)))  float        f32x4;
typedef unsigned short u16t;

#define BATCH 4
#define SEQ   4096
#define HID   512
#define QPAD  8                       // 16B row pad: avoids LDS bank conflicts
#define QROW  (HID + QPAD)

// ---------- bf16 helpers ----------
union BFU { __bf16 b; u16t u; };

__device__ __forceinline__ u16t f2bf(float x) {
  BFU c; c.b = (__bf16)x;          // native RNE convert
  return c.u;
}

union BF16x16 { u32x4 q[2]; u16t u[16]; __bf16 b[16]; v16bf v; };

__device__ __forceinline__ v16bf pack16(const float* t) {
  BF16x16 p;
#pragma unroll
  for (int i = 0; i < 16; ++i) p.b[i] = (__bf16)t[i];
  return p.v;
}

// ---------- fragment loaders ----------
// A-matrix (16x32 bf16): lane m = lane&15, g = lane>>4; elem e<8 -> k=k0+8g+e,
// e>=8 -> k=k0+16+8g+(e-8). Two contiguous 8-half (16B) runs.
__device__ __forceinline__ v16bf load_a_lds(const u16t* qs, int k0, int lane) {
  int m = lane & 15, g = lane >> 4;
  const u16t* r = qs + m * QROW + k0 + 8 * g;
  BF16x16 p;
  p.q[0] = *(const u32x4*)(r);
  p.q[1] = *(const u32x4*)(r + 16);
  return p.v;
}

// B-matrix (32x16 bf16, NT: source stored N x K row-major): lane n = lane&15,
// g = lane>>4; elem e<8 -> k = k0 + 8g + e, e>=8 -> k = k0 + 16 + 8g + (e-8). Two 8-half (16B) runs.
__device__ __forceinline__ v16bf load_b_bf(const u16t* base, int ld, int n0, int k0, int lane) {
  int n = n0 + (lane & 15), g = lane >> 4;
  const u16t* r = base + (size_t)n * ld + k0 + 8 * g;
  BF16x16 p;
  p.q[0] = *(const u32x4*)(r);
  p.q[1] = *(const u32x4*)(r + 16);
  return p.v;
}

__device__ __forceinline__ v16bf load_a_f32(const float* base, int ld, int m0, int k0, int lane) {
  int m = m0 + (lane & 15), g = lane >> 4;
  const float* r = base + (size_t)m * ld + k0 + 8 * g;
  f32x4 a = ((const f32x4*)r)[0];
  f32x4 b = ((const f32x4*)r)[1];
  f32x4 c = ((const f32x4*)(r + 16))[0];
  f32x4 d = ((const f32x4*)(r + 16))[1];
  float t[16];
#pragma unroll
  for (int i = 0; i < 4; ++i) { t[i] = a[i]; t[4 + i] = b[i]; t[8 + i] = c[i]; t[12 + i] = d[i]; }
  return pack16(t);
}

__device__ __forceinline__ v16bf load_b_f32(const float* base, int ld, int n0, int k0, int lane) {
  int n = n0 + (lane & 15), g = lane >> 4;
  const float* r = base + (size_t)n * ld + k0 + 8 * g;
  float t[16];
#pragma unroll
  for (int i = 0; i < 4; ++i) {
    f32x4 vv = ((const f32x4*)(r + (i >> 1) * 16))[i & 1];
#pragma unroll
    for (int j = 0; j < 4; ++j) t[4 * i + j] = vv[j];
  }
  return pack16(t);
}

__device__ __forceinline__ v8f wmma_bf16(v16bf a, v16bf b, v8f c) {
  return __builtin_amdgcn_wmma_f32_16x16x32_bf16(false, a, false, b, (short)0, c, false, false);
}

// ---------- kernel 1: projection GEMM  out = X @ W^T + bias (fp32 -> bf16) ----------
template <int TRANS>
__global__ void proj_kernel(const float* __restrict__ X, const float* __restrict__ W,
                            const float* __restrict__ bias, u16t* __restrict__ out,
                            int M, int N, int K, int out_ld) {
  int lane = threadIdx.x & 31;
  int wav  = blockIdx.x * (blockDim.x >> 5) + (threadIdx.x >> 5);
  int ntn  = N >> 4;
  if (wav >= (M >> 4) * ntn) return;
  int m0 = (wav / ntn) << 4, n0 = (wav % ntn) << 4;

  v8f acc = {};
  for (int k0 = 0; k0 < K; k0 += 32) {
    v16bf a = load_a_f32(X, K, m0, k0, lane);
    v16bf b = load_b_f32(W, K, n0, k0, lane);
    acc = wmma_bf16(a, b, acc);
  }
  int g = lane >> 4, n = n0 + (lane & 15);
  float bv = bias[n];
#pragma unroll
  for (int r = 0; r < 8; ++r) {
    int m = m0 + r + (g << 3);
    u16t v = f2bf(acc[r] + bv);
    if (TRANS) out[(size_t)n * out_ld + m] = v;
    else       out[(size_t)m * out_ld + n] = v;
  }
}

// ---------- kernel 2: causal softmax weights (two-pass online softmax) ----------
// Q tile staged in LDS (bank-conflict-free padded rows), K tiles streamed from L2.
__global__ void attn_weights_kernel(const u16t* __restrict__ Qb, const u16t* __restrict__ Kb,
                                    float* __restrict__ Wout) {
  extern __shared__ u16t smem[];               // 16*QROW bf16 per wave
  const float scale = 0.04419417382415922f;    // 1/sqrt(512)
  int lane = threadIdx.x & 31;
  int wv   = threadIdx.x >> 5;
  int wav  = blockIdx.x * (blockDim.x >> 5) + wv;
  int b = wav >> 8, qt = wav & 255;
  if (b >= BATCH) return;
  int m0 = qt << 4;
  const u16t* Qbase = Qb + (size_t)b * SEQ * HID;
  const u16t* Kbase = Kb + (size_t)b * SEQ * HID;
  float* Wb = Wout + (size_t)b * SEQ * SEQ;
  int g = lane >> 4, nloc = lane & 15;

  // stage this wave's Q tile with padded rows (16B pad -> 4-bank shift per row)
  u16t* qs = smem + wv * 16 * QROW;
  {
    const u32x4* src = (const u32x4*)(Qbase + (size_t)m0 * HID);
    for (int i = lane; i < 16 * (HID / 8); i += 32) {
      int row = i >> 6, chunk = i & 63;                 // 64 x 16B chunks per row
      ((u32x4*)(qs + row * QROW))[chunk] = src[(row << 6) + chunk];
    }
  }

  float rmax[8], rsum[8];
#pragma unroll
  for (int r = 0; r < 8; ++r) { rmax[r] = -3.0e38f; rsum[r] = 0.f; }

  // pass 1: running row max / sum
  for (int jt = 0; jt <= qt; ++jt) {
    if (jt + 1 <= qt)
      __builtin_prefetch(Kbase + (size_t)((jt + 1) << 4) * HID, 0, 1);
    int kbase = 0;
    asm volatile("" : "+v"(kbase));    // opaque: keep Q ds_loads inside the jt loop
    v8f acc = {};
#pragma unroll
    for (int t = 0; t < 16; ++t)
      acc = wmma_bf16(load_a_lds(qs, kbase + t * 32, lane),
                      load_b_bf(Kbase, HID, jt << 4, t * 32, lane), acc);
    int j = (jt << 4) + nloc;
#pragma unroll
    for (int r = 0; r < 8; ++r) {
      float s = acc[r] * scale;
      int i = m0 + r + (g << 3);
      if (j > i) s = -1.0e7f;            // only possible on the diagonal tile
      float tmax = s;
#pragma unroll
      for (int msk = 1; msk < 16; msk <<= 1) tmax = fmaxf(tmax, __shfl_xor(tmax, msk, 32));
      float nm = fmaxf(rmax[r], tmax);
      float p  = __expf(s - nm);
      float ts = p;
#pragma unroll
      for (int msk = 1; msk < 16; msk <<= 1) ts += __shfl_xor(ts, msk, 32);
      rsum[r] = rsum[r] * __expf(rmax[r] - nm) + ts;
      rmax[r] = nm;
    }
  }
  float rinv[8];
#pragma unroll
  for (int r = 0; r < 8; ++r) rinv[r] = 1.0f / rsum[r];

  // pass 2: recompute, normalize, store fp32 weights
  for (int jt = 0; jt <= qt; ++jt) {
    int kbase = 0;
    asm volatile("" : "+v"(kbase));
    v8f acc = {};
#pragma unroll
    for (int t = 0; t < 16; ++t)
      acc = wmma_bf16(load_a_lds(qs, kbase + t * 32, lane),
                      load_b_bf(Kbase, HID, jt << 4, t * 32, lane), acc);
    int j = (jt << 4) + nloc;
#pragma unroll
    for (int r = 0; r < 8; ++r) {
      float s = acc[r] * scale;
      int i = m0 + r + (g << 3);
      if (j > i) s = -1.0e7f;
      Wb[(size_t)i * SEQ + j] = __expf(s - rmax[r]) * rinv[r];
    }
  }

  // zero-fill above the causal boundary (required output) — b128 stores
  int jstart = (qt + 1) << 4;                  // multiple of 16 floats: 64B aligned
  f32x4 z = {};
  for (int m = 0; m < 16; ++m) {
    float* row = Wb + (size_t)(m0 + m) * SEQ;
    for (int j = jstart + lane * 4; j < SEQ; j += 128)
      *(f32x4*)(row + j) = z;
  }
}

// ---------- kernel 3: context = weights @ V  (A fp32->bf16, B = V^T bf16) ----------
__global__ void context_kernel(const float* __restrict__ Wf, const u16t* __restrict__ Vt,
                               float* __restrict__ Cout) {
  int lane = threadIdx.x & 31;
  int wav  = blockIdx.x * (blockDim.x >> 5) + (threadIdx.x >> 5);
  const int NT = HID >> 4, QT = SEQ >> 4;
  int b = wav / (QT * NT);
  if (b >= BATCH) return;
  int rem = wav % (QT * NT);
  int qt = rem / NT, nt = rem % NT;
  int m0 = qt << 4, n0 = nt << 4;
  const float* Wb = Wf + (size_t)b * SEQ * SEQ;
  const u16t*  Vb = Vt + (size_t)b * SEQ;        // V^T: [HID][BATCH*SEQ]
  int ksteps = (qt + 2) >> 1;                    // causal: K = 16*(qt+1), zeros pad K%32
  v8f acc = {};
  for (int ks = 0; ks < ksteps; ++ks) {
    v16bf a  = load_a_f32(Wb, SEQ, m0, ks * 32, lane);
    v16bf bm = load_b_bf(Vb, BATCH * SEQ, n0, ks * 32, lane);
    acc = wmma_bf16(a, bm, acc);
  }
  int g = lane >> 4, n = n0 + (lane & 15);
#pragma unroll
  for (int r = 0; r < 8; ++r) {
    int m = m0 + r + (g << 3);
    Cout[((size_t)b * SEQ + m) * HID + n] = acc[r];
  }
}

extern "C" void kernel_launch(void* const* d_in, const int* in_sizes, int n_in,
                              void* d_out, int out_size, void* d_ws, size_t ws_size,
                              hipStream_t stream) {
  const float* queries = (const float*)d_in[0];
  const float* keys    = (const float*)d_in[1];
  const float* values  = (const float*)d_in[2];
  const float* Wq = (const float*)d_in[3];
  const float* bq = (const float*)d_in[4];
  const float* Wk = (const float*)d_in[5];
  const float* bk = (const float*)d_in[6];
  const float* Wv = (const float*)d_in[7];
  const float* bv = (const float*)d_in[8];

  float* ctx  = (float*)d_out;                              // (B,S,H) fp32
  float* attw = ctx + (size_t)BATCH * SEQ * HID;            // (B,S,S) fp32

  u16t* Qb = (u16t*)d_ws;                                   // bf16 (B*S, H)
  u16t* Kb = Qb + (size_t)BATCH * SEQ * HID;                // bf16 (B*S, H)
  u16t* Vt = Kb + (size_t)BATCH * SEQ * HID;                // bf16 (H, B*S) transposed

  const int M = BATCH * SEQ;
  int tiles1  = (M >> 4) * (HID >> 4);                      // 32768
  proj_kernel<0><<<tiles1 / 8, 256, 0, stream>>>(queries, Wq, bq, Qb, M, HID, HID, HID);
  proj_kernel<0><<<tiles1 / 8, 256, 0, stream>>>(keys,    Wk, bk, Kb, M, HID, HID, HID);
  proj_kernel<1><<<tiles1 / 8, 256, 0, stream>>>(values,  Wv, bv, Vt, M, HID, HID, M);

  int waves2 = BATCH * (SEQ >> 4);                          // 1024
  size_t lds2 = (size_t)8 * 16 * QROW * sizeof(u16t);       // 130 KB
  attn_weights_kernel<<<waves2 / 8, 256, lds2, stream>>>(Qb, Kb, attw);

  int tiles3 = BATCH * (SEQ >> 4) * (HID >> 4);             // 32768
  context_kernel<<<tiles3 / 8, 256, 0, stream>>>(attw, Vt, ctx);
}
